// HGT_90572270338454
// MI455X (gfx1250) — matrix operand order repaired and checked by reading, not compiled, not run on hardware
//
#include <hip/hip_runtime.h>
#include <hip/hip_bf16.h>

#define HID   256
#define HEADS 8
#define DH    32
#define NG    30000
#define NP    6000
#define OUTD  64
#define E_GG  300000
#define E_GP  100000
#define E_PG  100000

typedef __attribute__((ext_vector_type(16))) __bf16 v16bf;
typedef __attribute__((ext_vector_type(8)))  __bf16 v8bf;
typedef __attribute__((ext_vector_type(8)))  float  v8f;

__device__ __forceinline__ __bf16 f2bf(float f) {
  unsigned u = __builtin_bit_cast(unsigned, f);
  u += 0x7fffu + ((u >> 16) & 1u);            // round-to-nearest-even
  unsigned short h = (unsigned short)(u >> 16);
  return __builtin_bit_cast(__bf16, h);
}

// ---------------------------------------------------------------------------
// WMMA bf16 GEMM: C[M,N] = epilogue(A[M,K](bf16) @ Wt[N,K]^T(bf16) + bias)
// One wave computes a 16x64 strip: one A fragment reused across 4 B fragments
// and 4 accumulators -> 4 back-to-back v_wmma per K-step, 4 independent
// dependency chains for the XDL pipeline.
// A fragment (16x32 bf16, ISA 7.12.2): lanes 0-15 row M=lm K={0..7,16..23},
// lanes 16-31 row M=lm K={8..15,24..31}. B fragment mirrors with N=lm.
// C/D: vgpr r -> M = half*8 + r, N = lm.
// ACT: 0=none, 1=relu.  SKIP: out = sig(gate)*o + (1-sig(gate))*Xskip.
// ---------------------------------------------------------------------------
template <int ACT, bool SKIP>
__global__ __launch_bounds__(128) void gemm_bf16(
    const __bf16* __restrict__ A, const __bf16* __restrict__ Wt,
    const float* __restrict__ bias, float* __restrict__ C,
    __bf16* __restrict__ Cb, const float* __restrict__ Xskip,
    const float* __restrict__ gate, int M, int N, int K) {
  const int lane = threadIdx.x;
  const int half = lane >> 4, lm = lane & 15;
  const int tm = blockIdx.x;
  const int colBase = (blockIdx.y * blockDim.y + threadIdx.y) * 64;
  const int row = tm * 16 + lm;   // A row held by this lane
  const __bf16* __restrict__ arow = A + (size_t)row * K;
  const __bf16* __restrict__ brow0 = Wt + (size_t)(colBase + lm) * K;
  const size_t bstride = (size_t)16 * K;
  v8f acc[4] = {{}, {}, {}, {}};
  for (int k0 = 0; k0 < K; k0 += 32) {
    __builtin_prefetch(arow + k0 + 128, 0, 1);   // global_prefetch_b8
    v8bf alo = *(const v8bf*)(arow + k0 + half * 8);
    v8bf ahi = *(const v8bf*)(arow + k0 + 16 + half * 8);
    v16bf a;
#pragma unroll
    for (int i = 0; i < 8; ++i) { a[i] = alo[i]; a[i + 8] = ahi[i]; }
#pragma unroll
    for (int j = 0; j < 4; ++j) {
      const __bf16* bp = brow0 + j * bstride + k0 + half * 8;
      v8bf blo = *(const v8bf*)(bp);
      v8bf bhi = *(const v8bf*)(bp + 16);
      v16bf b;
#pragma unroll
      for (int i = 0; i < 8; ++i) { b[i] = blo[i]; b[i + 8] = bhi[i]; }
      acc[j] = __builtin_amdgcn_wmma_f32_16x16x32_bf16(false, a, false, b,
                                                       (short)0, acc[j], false, false);
    }
  }
  float g = 0.0f, og = 0.0f;
  if (SKIP) { g = 1.0f / (1.0f + __expf(-gate[0])); og = 1.0f - g; }
  const int m0 = tm * 16 + half * 8;
#pragma unroll
  for (int j = 0; j < 4; ++j) {
    const int col = colBase + j * 16 + lm;
    const float bc = bias ? bias[col] : 0.0f;
#pragma unroll
    for (int r = 0; r < 8; ++r) {
      const size_t idx = (size_t)(m0 + r) * N + col;
      float v = acc[j][r] + bc;
      if (ACT == 1) v = fmaxf(v, 0.0f);
      if (SKIP) v = g * v + og * Xskip[idx];
      if (C) C[idx] = v;
      if (Cb) Cb[idx] = f2bf(v);
    }
  }
}

// ---------------------------------------------------------------------------
// Per-head relation transform via WMMA: out[n, h*32+e] = sum_d A[n,h*32+d]*rel[h,d,e]
// A: bf16 [n, 256] (k or v mirror). relT: bf16 [8][32(e)][32(d)] (transposed).
// grid = (n/16, HEADS), block = (32, 2): wave ty covers 16 of the 32 head cols.
// Single 16x16x32 WMMA K-step per tile.
// ---------------------------------------------------------------------------
__global__ __launch_bounds__(64) void k_rel_wmma(
    const __bf16* __restrict__ A, const __bf16* __restrict__ relT,
    float* __restrict__ out, int n) {
  const int lane = threadIdx.x;
  const int half = lane >> 4, lm = lane & 15;
  const int h = blockIdx.y;
  const int row = blockIdx.x * 16 + lm;
  const int colInHead = threadIdx.y * 16 + lm;
  const __bf16* __restrict__ arow = A + (size_t)row * HID + h * DH;
  const __bf16* __restrict__ brow = relT + h * DH * DH + (size_t)colInHead * DH;
  v8bf alo = *(const v8bf*)(arow + half * 8);
  v8bf ahi = *(const v8bf*)(arow + 16 + half * 8);
  v8bf blo = *(const v8bf*)(brow + half * 8);
  v8bf bhi = *(const v8bf*)(brow + 16 + half * 8);
  v16bf a, b;
#pragma unroll
  for (int i = 0; i < 8; ++i) {
    a[i] = alo[i]; a[i + 8] = ahi[i];
    b[i] = blo[i]; b[i + 8] = bhi[i];
  }
  v8f acc = {};
  acc = __builtin_amdgcn_wmma_f32_16x16x32_bf16(false, a, false, b,
                                                (short)0, acc, false, false);
  const int m0 = blockIdx.x * 16 + half * 8;
#pragma unroll
  for (int r = 0; r < 8; ++r)
    out[(size_t)(m0 + r) * HID + h * DH + colInHead] = acc[r];
}

// rel [8,32(d),32(e)] f32 -> relT [8,32(e),32(d)] bf16
__global__ void k_rel_prep(const float* __restrict__ rel, __bf16* __restrict__ relT) {
  int i = blockIdx.x * blockDim.x + threadIdx.x;
  if (i >= HEADS * DH * DH) return;
  int h = i >> 10, rem = i & 1023, e = rem >> 5, d = rem & 31;
  relT[i] = f2bf(rel[(h << 10) + (d << 5) + e]);
}

// ---------------------------------------------------------------------------
// Small support kernels
// ---------------------------------------------------------------------------
__global__ void k_f32_to_bf16(const float* __restrict__ x, __bf16* __restrict__ y, int n) {
  int i = blockIdx.x * blockDim.x + threadIdx.x;
  if (i < n) y[i] = f2bf(x[i]);
}

__global__ void k_gelu_to_bf16(const float* __restrict__ x, __bf16* __restrict__ y, int n) {
  int i = blockIdx.x * blockDim.x + threadIdx.x;
  if (i < n) {
    float v = x[i];
    y[i] = f2bf(0.5f * v * (1.0f + erff(v * 0.70710678118f)));
  }
}

// W[K,N] f32 -> Wt[N,K] bf16
__global__ void k_transpose_w(const float* __restrict__ W, __bf16* __restrict__ Wt,
                              int K, int N) {
  int i = blockIdx.x * blockDim.x + threadIdx.x;
  if (i >= K * N) return;
  int n = i / K, k = i - n * K;
  Wt[i] = f2bf(W[(size_t)k * N + n]);
}

__global__ void k_fill_f32(float* __restrict__ p, float v, int n) {
  int i = blockIdx.x * blockDim.x + threadIdx.x;
  if (i < n) p[i] = v;
}
__global__ void k_fill_u32(unsigned* __restrict__ p, unsigned v, int n) {
  int i = blockIdx.x * blockDim.x + threadIdx.x;
  if (i < n) p[i] = v;
}

// pass 1: raw scores + per-(dst,head) running max via order-preserving uint
__global__ void k_edge_scores(const int* __restrict__ src, const int* __restrict__ dst,
                              int E, const float* __restrict__ q, const float* __restrict__ kk,
                              const float* __restrict__ prel, float scale,
                              float* __restrict__ alpha, unsigned* __restrict__ dmax) {
  int t = blockIdx.x * blockDim.x + threadIdx.x;
  if (t >= E * HEADS) return;
  int e = t >> 3, h = t & 7;
  int s = src[e], d = dst[e];
  const float* qp = q + (size_t)d * HID + h * DH;
  const float* kp = kk + (size_t)s * HID + h * DH;
  float acc = 0.0f;
#pragma unroll
  for (int i = 0; i < DH; ++i) acc = fmaf(qp[i], kp[i], acc);
  acc *= prel[h] * scale;
  alpha[t] = acc;
  unsigned u = __builtin_bit_cast(unsigned, acc);
  u = (u & 0x80000000u) ? ~u : (u | 0x80000000u);
  atomicMax(&dmax[(size_t)d * HEADS + h], u);
}

// pass 2: exp(score - max), accumulate per-(dst,head) sums
__global__ void k_edge_expsum(const int* __restrict__ dst, int E,
                              float* __restrict__ alpha,
                              const unsigned* __restrict__ dmax,
                              float* __restrict__ dsum) {
  int t = blockIdx.x * blockDim.x + threadIdx.x;
  if (t >= E * HEADS) return;
  int e = t >> 3, h = t & 7;
  int d = dst[e];
  unsigned u = dmax[(size_t)d * HEADS + h];
  float m = __builtin_bit_cast(float, (u & 0x80000000u) ? (u ^ 0x80000000u) : ~u);
  float w = __expf(alpha[t] - m);
  alpha[t] = w;
  atomicAdd(&dsum[(size_t)d * HEADS + h], w);
}

// pass 3: agg[dst, c] += v[src, c] * alpha / (sum + eps); one thread per (edge, c)
__global__ void k_edge_agg(const int* __restrict__ src, const int* __restrict__ dst, int E,
                           const float* __restrict__ v, const float* __restrict__ alpha,
                           const float* __restrict__ dsum, float* __restrict__ agg) {
  int t = blockIdx.x * blockDim.x + threadIdx.x;
  if (t >= E * HID) return;
  int e = t >> 8, c = t & 255;
  int h = c >> 5;
  int s = src[e], d = dst[e];
  float w = alpha[(size_t)e * HEADS + h] / (dsum[(size_t)d * HEADS + h] + 1e-16f);
  atomicAdd(&agg[(size_t)d * HID + c], v[(size_t)s * HID + c] * w);
}

// ---------------------------------------------------------------------------
// Host side
// ---------------------------------------------------------------------------
static inline int cdiv(int a, int b) { return (a + b - 1) / b; }

static void launch_gemm(const __bf16* A, const __bf16* Wt, const float* bias,
                        float* C, __bf16* Cb, const float* Xskip, const float* gate,
                        int M, int N, int K, int act, hipStream_t s) {
  int wavesN = N / 64;                 // each wave covers 64 output columns
  int by = wavesN >= 4 ? 4 : wavesN;
  dim3 grid(M / 16, wavesN / by), block(32, by);
  if (Xskip)
    gemm_bf16<0, true><<<grid, block, 0, s>>>(A, Wt, bias, C, Cb, Xskip, gate, M, N, K);
  else if (act == 1)
    gemm_bf16<1, false><<<grid, block, 0, s>>>(A, Wt, bias, C, Cb, nullptr, nullptr, M, N, K);
  else
    gemm_bf16<0, false><<<grid, block, 0, s>>>(A, Wt, bias, C, Cb, nullptr, nullptr, M, N, K);
}

extern "C" void kernel_launch(void* const* d_in, const int* in_sizes, int n_in,
                              void* d_out, int out_size, void* d_ws, size_t ws_size,
                              hipStream_t stream) {
  (void)in_sizes; (void)n_in; (void)out_size; (void)ws_size;
  // d_in layout: insertion order for top-level inputs, JAX pytree leaf order
  // (dict keys sorted) inside params.
  const float* x_gene = (const float*)d_in[0];
  const float* x_path = (const float*)d_in[1];
  const int* ei[3] = {(const int*)d_in[2], (const int*)d_in[3], (const int*)d_in[4]};
  const float* dec_w   = (const float*)d_in[5];
  const float* dec_b   = (const float*)d_in[6];
  const float* enc_g_w = (const float*)d_in[7];
  const float* enc_g_b = (const float*)d_in[8];
  const float* enc_p_w = (const float*)d_in[9];
  const float* enc_p_b = (const float*)d_in[10];

  struct Lp {
    const float *a_w[2], *a_b[2];      // a: gene, pathway
    const float *arel[3], *mrel[3], *prel[3];
    const float *k_w[2], *k_b[2];
    const float *q_w[2], *q_b[2];
    const float *skip[2];
    const float *v_w[2], *v_b[2];
  } L[2];
  for (int l = 0; l < 2; ++l) {
    int b = 11 + 27 * l;
    L[l].a_w[0] = (const float*)d_in[b + 0];  L[l].a_b[0] = (const float*)d_in[b + 1];
    L[l].a_w[1] = (const float*)d_in[b + 2];  L[l].a_b[1] = (const float*)d_in[b + 3];
    for (int e = 0; e < 3; ++e) L[l].arel[e] = (const float*)d_in[b + 4 + e];
    L[l].k_w[0] = (const float*)d_in[b + 7];  L[l].k_b[0] = (const float*)d_in[b + 8];
    L[l].k_w[1] = (const float*)d_in[b + 9];  L[l].k_b[1] = (const float*)d_in[b + 10];
    for (int e = 0; e < 3; ++e) L[l].mrel[e] = (const float*)d_in[b + 11 + e];
    for (int e = 0; e < 3; ++e) L[l].prel[e] = (const float*)d_in[b + 14 + e];
    L[l].q_w[0] = (const float*)d_in[b + 17]; L[l].q_b[0] = (const float*)d_in[b + 18];
    L[l].q_w[1] = (const float*)d_in[b + 19]; L[l].q_b[1] = (const float*)d_in[b + 20];
    L[l].skip[0] = (const float*)d_in[b + 21]; L[l].skip[1] = (const float*)d_in[b + 22];
    L[l].v_w[0] = (const float*)d_in[b + 23]; L[l].v_b[0] = (const float*)d_in[b + 24];
    L[l].v_w[1] = (const float*)d_in[b + 25]; L[l].v_b[1] = (const float*)d_in[b + 26];
  }

  // ---- workspace carve ----
  char* ws = (char*)d_ws;
  size_t off = 0;
  auto carve = [&](size_t bytes) -> char* {
    char* p = ws + off;
    off += (bytes + 255) & ~(size_t)255;
    return p;
  };
  float*  xg   = (float*)carve((size_t)NG * HID * 4);
  float*  xp   = (float*)carve((size_t)NP * HID * 4);
  float*  xg2  = (float*)carve((size_t)NG * HID * 4);
  float*  xp2  = (float*)carve((size_t)NP * HID * 4);
  __bf16* xgb  = (__bf16*)carve((size_t)NG * HID * 2);
  __bf16* xpb  = (__bf16*)carve((size_t)NP * HID * 2);
  __bf16* xginb = (__bf16*)carve((size_t)NG * 256 * 2);
  __bf16* xpinb = (__bf16*)carve((size_t)NP * 128 * 2);
  __bf16* kb[2]; __bf16* vb[2];        // bf16 k / v mirrors per node type
  float*  qf[2];                        // f32 q per node type
  for (int t = 0; t < 2; ++t) {
    size_t nNodes = (t == 0) ? NG : NP;
    kb[t] = (__bf16*)carve(nNodes * HID * 2);
    vb[t] = (__bf16*)carve(nNodes * HID * 2);
    qf[t] = (float*)carve(nNodes * HID * 4);
  }
  float*  krel = (float*)carve((size_t)NG * HID * 4);
  float*  vrel = (float*)carve((size_t)NG * HID * 4);
  float*  alpha = (float*)carve((size_t)E_GG * HEADS * 4);
  unsigned* dmax = (unsigned*)carve((size_t)NG * HEADS * 4);
  float*  dsum = (float*)carve((size_t)NG * HEADS * 4);
  float*  aggg = (float*)carve((size_t)NG * HID * 4);
  float*  aggp = (float*)carve((size_t)NP * HID * 4);
  __bf16* aggb = (__bf16*)carve((size_t)NG * HID * 2);
  __bf16* arelT = (__bf16*)carve((size_t)HEADS * DH * DH * 2);
  __bf16* mrelT = (__bf16*)carve((size_t)HEADS * DH * DH * 2);
  __bf16* wt_enc_g = (__bf16*)carve((size_t)256 * 256 * 2);
  __bf16* wt_enc_p = (__bf16*)carve((size_t)256 * 128 * 2);
  __bf16* wt_dec   = (__bf16*)carve((size_t)64 * 256 * 2);
  __bf16 *wt_k[2][2], *wt_q[2][2], *wt_v[2][2], *wt_a[2][2];
  for (int l = 0; l < 2; ++l)
    for (int t = 0; t < 2; ++t) {
      wt_k[l][t] = (__bf16*)carve((size_t)256 * 256 * 2);
      wt_q[l][t] = (__bf16*)carve((size_t)256 * 256 * 2);
      wt_v[l][t] = (__bf16*)carve((size_t)256 * 256 * 2);
      wt_a[l][t] = (__bf16*)carve((size_t)256 * 256 * 2);
    }

  // ---- weight prep (bf16 transposed) ----
  k_transpose_w<<<cdiv(256 * 256, 256), 256, 0, stream>>>(enc_g_w, wt_enc_g, 256, 256);
  k_transpose_w<<<cdiv(128 * 256, 256), 256, 0, stream>>>(enc_p_w, wt_enc_p, 128, 256);
  k_transpose_w<<<cdiv(256 * 64, 256), 256, 0, stream>>>(dec_w, wt_dec, 256, 64);
  for (int l = 0; l < 2; ++l)
    for (int t = 0; t < 2; ++t) {
      k_transpose_w<<<cdiv(256 * 256, 256), 256, 0, stream>>>(L[l].k_w[t], wt_k[l][t], 256, 256);
      k_transpose_w<<<cdiv(256 * 256, 256), 256, 0, stream>>>(L[l].q_w[t], wt_q[l][t], 256, 256);
      k_transpose_w<<<cdiv(256 * 256, 256), 256, 0, stream>>>(L[l].v_w[t], wt_v[l][t], 256, 256);
      k_transpose_w<<<cdiv(256 * 256, 256), 256, 0, stream>>>(L[l].a_w[t], wt_a[l][t], 256, 256);
    }

  // ---- encoders: relu(x @ W + b), f32 + bf16 mirror ----
  k_f32_to_bf16<<<cdiv(NG * 256, 256), 256, 0, stream>>>(x_gene, xginb, NG * 256);
  k_f32_to_bf16<<<cdiv(NP * 128, 256), 256, 0, stream>>>(x_path, xpinb, NP * 128);
  launch_gemm(xginb, wt_enc_g, enc_g_b, xg, xgb, nullptr, nullptr, NG, HID, 256, 1, stream);
  launch_gemm(xpinb, wt_enc_p, enc_p_b, xp, xpb, nullptr, nullptr, NP, HID, 128, 1, stream);

  // edge type table: {src type, dst type} 0=gene 1=pathway
  const int etSrc[3] = {0, 0, 1}, etDst[3] = {0, 1, 0};
  const int etE[3] = {E_GG, E_GP, E_PG};
  const float scale = 0.17677669529f;  // 1/sqrt(32)

  float *cg = xg, *cp = xp, *nxg = xg2, *nxp = xp2;
  for (int l = 0; l < 2; ++l) {
    const __bf16* xb[2] = {xgb, xpb};
    const int nNodes[2] = {NG, NP};
    // k / q / v linears (WMMA): k,v -> bf16 mirrors only; q -> f32
    for (int t = 0; t < 2; ++t) {
      launch_gemm(xb[t], wt_k[l][t], L[l].k_b[t], nullptr, kb[t], nullptr, nullptr,
                  nNodes[t], HID, HID, 0, stream);
      launch_gemm(xb[t], wt_q[l][t], L[l].q_b[t], qf[t], nullptr, nullptr, nullptr,
                  nNodes[t], HID, HID, 0, stream);
      launch_gemm(xb[t], wt_v[l][t], L[l].v_b[t], nullptr, vb[t], nullptr, nullptr,
                  nNodes[t], HID, HID, 0, stream);
    }
    k_fill_f32<<<cdiv(NG * HID, 256), 256, 0, stream>>>(aggg, 0.0f, NG * HID);
    k_fill_f32<<<cdiv(NP * HID, 256), 256, 0, stream>>>(aggp, 0.0f, NP * HID);

    for (int e = 0; e < 3; ++e) {
      const int st = etSrc[e], dt = etDst[e], E = etE[e];
      const int nsrc = nNodes[st], ndst = nNodes[dt];
      const float* qdst = qf[dt];
      float* agg = (dt == 0) ? aggg : aggp;
      const int* src = ei[e];
      const int* dst = ei[e] + E;
      // relation transforms via WMMA (one 16x16x32 matrix op per tile)
      k_rel_prep<<<cdiv(HEADS * DH * DH, 256), 256, 0, stream>>>(L[l].arel[e], arelT);
      k_rel_prep<<<cdiv(HEADS * DH * DH, 256), 256, 0, stream>>>(L[l].mrel[e], mrelT);
      k_rel_wmma<<<dim3(nsrc / 16, HEADS), dim3(32, 2), 0, stream>>>(kb[st], arelT, krel, nsrc);
      k_rel_wmma<<<dim3(nsrc / 16, HEADS), dim3(32, 2), 0, stream>>>(vb[st], mrelT, vrel, nsrc);
      k_fill_u32<<<cdiv(ndst * HEADS, 256), 256, 0, stream>>>(dmax, 0x007FFFFFu, ndst * HEADS); // enc(-inf)
      k_fill_f32<<<cdiv(ndst * HEADS, 256), 256, 0, stream>>>(dsum, 0.0f, ndst * HEADS);
      k_edge_scores<<<cdiv(E * HEADS, 256), 256, 0, stream>>>(src, dst, E, qdst, krel,
                                                              L[l].prel[e], scale, alpha, dmax);
      k_edge_expsum<<<cdiv(E * HEADS, 256), 256, 0, stream>>>(dst, E, alpha, dmax, dsum);
      k_edge_agg<<<cdiv(E * HID, 256), 256, 0, stream>>>(src, dst, E, vrel, alpha, dsum, agg);
    }

    // out = sig(skip)*lin(gelu(agg)) + (1-sig)*x   (WMMA + fused epilogue)
    k_gelu_to_bf16<<<cdiv(NG * HID, 256), 256, 0, stream>>>(aggg, aggb, NG * HID);
    launch_gemm(aggb, wt_a[l][0], L[l].a_b[0], nxg, xgb, cg, L[l].skip[0], NG, HID, HID, 0, stream);
    k_gelu_to_bf16<<<cdiv(NP * HID, 256), 256, 0, stream>>>(aggp, aggb, NP * HID);
    launch_gemm(aggb, wt_a[l][1], L[l].a_b[1], nxp, xpb, cp, L[l].skip[1], NP, HID, HID, 0, stream);
    float* tmp;
    tmp = cg; cg = nxg; nxg = tmp;
    tmp = cp; cp = nxp; nxp = tmp;
  }

  // ---- decoder: writes both outputs flat into d_out (gene then pathway) ----
  float* out = (float*)d_out;
  launch_gemm(xgb, wt_dec, dec_b, out, nullptr, nullptr, nullptr, NG, OUTD, HID, 0, stream);
  launch_gemm(xpb, wt_dec, dec_b, out + (size_t)NG * OUTD, nullptr, nullptr, nullptr,
              NP, OUTD, HID, 0, stream);
}
